// EnhancedGIN_20048907337871
// MI455X (gfx1250) — compile-verified, matrix operand order checked by reading, not run on hardware
//
#include <hip/hip_runtime.h>
#include <hip/hip_bf16.h>
#include <cstdint>

// ---------------------------------------------------------------------------
// EnhancedGIN forward for MI455X (gfx1250, wave32).
// GEMMs run as bf16 WMMA (v_wmma_f32_16x16x32_bf16). The 32KB weight matrix is
// staged pre-transposed into LDS via gfx1250 async-to-LDS copies
// (global_load_async_to_lds_b128 + s_wait_asynccnt). Per K-step all 8 B
// fragments are loaded into distinct registers first (one ds_load_b128 clause)
// so the 8 WMMAs issue back-to-back instead of each waiting dscnt==0.
// A fragments are global_load_b128 per the documented 16-bit A layout.
// Scatter-add / pooling use f32 global atomics. fc1_b/fc2_b are dropped: a
// batch-stat BN immediately follows each linear, so per-column biases cancel.
// M is padded to 50048 (50000 % 16 == 0, so tiles never mix real & pad rows;
// pad garbage is row-contained and never reaches real outputs).
// ---------------------------------------------------------------------------

#define HID        128
#define LAT        64
#define NNODES     50000
#define NEDGES     600000
#define NGRAPHS    512
#define NLAYERS    4
#define MPAD       50048           // ceil(50000/128)*128
#define STAT_CHUNK 512

typedef __bf16 bf16_t;
typedef __attribute__((ext_vector_type(16))) __bf16        v16bf;
typedef __attribute__((ext_vector_type(8)))  float         v8f;
typedef __attribute__((ext_vector_type(4)))  unsigned int  uint4v;

// ---------------------------------------------------------------------------
// Generic zero
// ---------------------------------------------------------------------------
__global__ void zero_f32_kernel(float* __restrict__ p, int n) {
    int t = blockIdx.x * 256 + threadIdx.x;
    if (t < n) p[t] = 0.0f;
}

// ---------------------------------------------------------------------------
// Convert weights [nMats][K=128][N=128] f32 -> transposed bf16 [nMats][N][K]
// ---------------------------------------------------------------------------
__global__ void wconv_kernel(const float* __restrict__ w, bf16_t* __restrict__ wt, int nMats) {
    int t = blockIdx.x * 256 + threadIdx.x;
    if (t >= nMats * HID * HID) return;
    int m   = t / (HID * HID);
    int rem = t - m * HID * HID;
    int n   = rem / HID;
    int k   = rem - n * HID;
    wt[t] = (bf16_t)w[(size_t)m * HID * HID + (size_t)k * HID + n];
}

// ---------------------------------------------------------------------------
// Column stats (sum, sumsq) over nRows rows; stats must be pre-zeroed.
// ---------------------------------------------------------------------------
__global__ void colstats_kernel(const float* __restrict__ X, int nRows,
                                float* __restrict__ stats) {
    int c  = threadIdx.x;                     // 128 threads
    int r0 = blockIdx.x * STAT_CHUNK;
    int r1 = r0 + STAT_CHUNK; if (r1 > nRows) r1 = nRows;
    float s = 0.f, s2 = 0.f;
    for (int r = r0; r < r1; ++r) {
        float v = X[(size_t)r * HID + c];
        s += v; s2 += v * v;
    }
    atomicAdd(&stats[c],        s);
    atomicAdd(&stats[HID + c],  s2);
}

__global__ void bn_finalize_kernel(const float* __restrict__ stats,
                                   const float* __restrict__ g,
                                   const float* __restrict__ b,
                                   float invN, float* __restrict__ ss) {
    int c = threadIdx.x;                      // 128 threads
    float mean = stats[c] * invN;
    float var  = stats[HID + c] * invN - mean * mean;
    float sc   = g[c] * rsqrtf(var + 1e-5f);
    ss[c]       = sc;
    ss[HID + c] = b[c] - mean * sc;
}

// ---------------------------------------------------------------------------
// Elementwise kernels over real rows only (pad rows contained by design)
// ---------------------------------------------------------------------------
__global__ void apply_input_bn_kernel(const float* __restrict__ x,
                                      const float* __restrict__ ss,
                                      float* __restrict__ xcur, int nElem) {
    int t = blockIdx.x * 256 + threadIdx.x;
    if (t >= nElem) return;
    int c = t & (HID - 1);
    xcur[t] = x[t] * ss[c] + ss[HID + c];
}

__global__ void hin_kernel(const float* __restrict__ xcur,
                           const float* __restrict__ agg,
                           const float* __restrict__ epsArr, int l,
                           bf16_t* __restrict__ abuf, int nElem) {
    int t = blockIdx.x * 256 + threadIdx.x;
    if (t >= nElem) return;
    abuf[t] = (bf16_t)((1.0f + epsArr[l]) * xcur[t] + agg[t]);
}

__device__ __forceinline__ float gelu_exact(float v) {
    return 0.5f * v * (1.0f + erff(v * 0.70710678118654752f));
}

// bn + gelu; optional bf16 out (next GEMM input), f32 out (x_cur), bf16 hs out
__global__ void bn_gelu_kernel(const float* __restrict__ y,
                               const float* __restrict__ ss,
                               bf16_t* __restrict__ obf,
                               float*  __restrict__ of32,
                               bf16_t* __restrict__ ohs, int nElem) {
    int t = blockIdx.x * 256 + threadIdx.x;
    if (t >= nElem) return;
    int c = t & (HID - 1);
    float v  = y[t] * ss[c] + ss[HID + c];
    float ge = gelu_exact(v);
    if (obf)  obf[t]  = (bf16_t)ge;
    if (of32) of32[t] = ge;
    if (ohs)  ohs[t]  = (bf16_t)ge;
}

// ---------------------------------------------------------------------------
// Edge scatter: one wave (32 lanes) per edge, float4 per lane, f32 atomics
// ---------------------------------------------------------------------------
__global__ void scatter_add_kernel(const float* __restrict__ x,
                                   const int* __restrict__ src,
                                   const int* __restrict__ dst,
                                   float* __restrict__ agg, int nEdges) {
    long long tid = (long long)blockIdx.x * 256 + threadIdx.x;
    int e = (int)(tid >> 5);
    if (e >= nEdges) return;
    int lane = (int)(tid & 31);
    int s = src[e], d = dst[e];
    const float* xs = x   + (size_t)s * HID + lane * 4;
    float*       ad = agg + (size_t)d * HID + lane * 4;
    float4 v = *(const float4*)xs;
    atomicAdd(ad + 0, v.x);
    atomicAdd(ad + 1, v.y);
    atomicAdd(ad + 2, v.z);
    atomicAdd(ad + 3, v.w);
}

// ---------------------------------------------------------------------------
// WMMA GEMM: Y[Mpad,128] = A[Mpad,128](bf16) @ B[128,128], B given transposed
// bf16 Bt[n][k]. 256 threads = 8 waves; wave w -> rows [blk*128+w*16, +16),
// 8 N-tiles, K unrolled 4x32 -> 32 v_wmma_f32_16x16x32_bf16 per block pass.
// B staged to LDS with gfx1250 async-to-LDS (ASYNCcnt) copies; all 8 B
// fragments of a K-step are loaded (distinct registers, one ds clause) before
// the 8 back-to-back WMMAs.
// ---------------------------------------------------------------------------
union BFrag { uint4v u[2]; v16bf v; };

__global__ __launch_bounds__(256)
void wmma_gemm_kernel(const bf16_t* __restrict__ A,
                      const bf16_t* __restrict__ Bt,
                      float* __restrict__ Y) {
    __shared__ __attribute__((aligned(16))) bf16_t bs[HID * HID];   // 32 KB

    int tid = threadIdx.x;
    // Async-stage transposed B into LDS: 2048 x b128 total, 8 per thread.
    // global_load_async_to_lds_b128: VDST = LDS byte address (addr[31:0] of the
    // flat shared pointer per the aperture rules), VADDR = 64-bit global addr.
    {
        #pragma unroll
        for (int i = 0; i < (HID * HID) / (8 * 256); ++i) {
            int idx = (tid + i * 256) * 8;          // 8 bf16 = 16 bytes
            unsigned      lds = (unsigned)(uintptr_t)(bs + idx);
            const bf16_t* g   = Bt + idx;
            asm volatile("global_load_async_to_lds_b128 %0, %1, off"
                         :: "v"(lds), "v"(g) : "memory");
        }
        asm volatile("s_wait_asynccnt 0x0" ::: "memory");
    }
    __syncthreads();

    int wave = tid >> 5;
    int lane = tid & 31;
    int half = lane >> 4;          // 0: lanes 0-15, 1: lanes 16-31
    int lid  = lane & 15;
    int m0   = blockIdx.x * 128 + wave * 16;

    v8f zero = {};
    v8f acc[8];
    #pragma unroll
    for (int t = 0; t < 8; ++t) acc[t] = zero;

    const bf16_t* arow = A + (size_t)(m0 + lid) * HID;

    #pragma unroll
    for (int kk = 0; kk < 4; ++kk) {
        const int k0 = kk * 32;
        // A fragment (16-bit A layout: lane<16 -> K{0..7,16..23}, lane>=16 -> K{8..15,24..31})
        BFrag af;
        af.u[0] = *(const uint4v*)(arow + k0 + 8 * half);
        af.u[1] = *(const uint4v*)(arow + k0 + 16 + 8 * half);

        // Load all 8 B fragments first: overlapping live ranges force distinct
        // registers -> one ds_load clause, then 8 back-to-back WMMAs.
        BFrag bfr[8];
        #pragma unroll
        for (int t = 0; t < 8; ++t) {
            const bf16_t* brow = bs + (size_t)(t * 16 + lid) * HID + k0;
            bfr[t].u[0] = *(const uint4v*)(brow + 8 * half);
            bfr[t].u[1] = *(const uint4v*)(brow + 16 + 8 * half);
        }
        #pragma unroll
        for (int t = 0; t < 8; ++t) {
            acc[t] = __builtin_amdgcn_wmma_f32_16x16x32_bf16(
                false, af.v, false, bfr[t].v, (short)0, acc[t], false, false);
        }
    }

    // C/D layout: VGPR r -> row m0 + r + 8*half, col = tile*16 + lid
    #pragma unroll
    for (int t = 0; t < 8; ++t) {
        int n = t * 16 + lid;
        #pragma unroll
        for (int r = 0; r < 8; ++r) {
            int m = m0 + r + 8 * half;
            Y[(size_t)m * HID + n] = acc[t][r];
        }
    }
}

// ---------------------------------------------------------------------------
// Pooling init / JK attention + fused pooling / finalize
// ---------------------------------------------------------------------------
__global__ void pool_init_kernel(float* __restrict__ sums, float* __restrict__ mx,
                                 float* __restrict__ cnt) {
    int t = blockIdx.x * 256 + threadIdx.x;
    if (t < NGRAPHS * HID) { sums[t] = 0.0f; mx[t] = -__builtin_inff(); }
    if (t < NGRAPHS) cnt[t] = 0.0f;
}

__device__ __forceinline__ void atomicMaxFloat(float* addr, float value) {
    if (value >= 0.0f)
        atomicMax((int*)addr, __float_as_int(value));
    else
        atomicMin((unsigned int*)addr, __float_as_uint(value));
}

__global__ __launch_bounds__(256)
void jk_pool_kernel(const bf16_t* __restrict__ hs, const float* __restrict__ attw,
                    const int* __restrict__ batch, float* __restrict__ sums,
                    float* __restrict__ mx, float* __restrict__ cnt) {
    int wave = threadIdx.x >> 5;
    int lane = threadIdx.x & 31;
    int n = blockIdx.x * 8 + wave;
    if (n >= NNODES) return;                 // wave-uniform exit
    int c0 = lane * 4;

    float hv[NLAYERS][4];
    float sc[NLAYERS];
    #pragma unroll
    for (int l = 0; l < NLAYERS; ++l) {
        const bf16_t* p = hs + ((size_t)l * MPAD + n) * HID + c0;
        float part = 0.0f;
        #pragma unroll
        for (int j = 0; j < 4; ++j) {
            float v = (float)p[j];
            hv[l][j] = v;
            part += v * attw[l * HID + c0 + j];
        }
        #pragma unroll
        for (int off = 16; off >= 1; off >>= 1)
            part += __shfl_xor(part, off, 32);
        sc[l] = part * (1.0f / HID);
    }
    // softmax over the 4 layers (all lanes have full sums)
    float mv = fmaxf(fmaxf(sc[0], sc[1]), fmaxf(sc[2], sc[3]));
    float e[NLAYERS], esum = 0.0f;
    #pragma unroll
    for (int l = 0; l < NLAYERS; ++l) { e[l] = expf(sc[l] - mv); esum += e[l]; }
    float inv = 1.0f / esum;

    int g = batch[n];
    float* sgp = sums + (size_t)g * HID + c0;
    float* mgp = mx   + (size_t)g * HID + c0;
    #pragma unroll
    for (int j = 0; j < 4; ++j) {
        float xv = 0.0f;
        #pragma unroll
        for (int l = 0; l < NLAYERS; ++l) xv += hv[l][j] * e[l];
        xv *= inv;
        atomicAdd(sgp + j, xv);
        atomicMaxFloat(mgp + j, xv);
    }
    if (lane == 0) atomicAdd(&cnt[g], 1.0f);
}

__global__ void pool_finalize_kernel(const float* __restrict__ sums,
                                     const float* __restrict__ mx,
                                     const float* __restrict__ cnt,
                                     const float* __restrict__ pw3,
                                     float* __restrict__ pooled) {
    int t = blockIdx.x * 256 + threadIdx.x;
    if (t >= NGRAPHS * HID) return;
    float w0 = pw3[0], w1 = pw3[1], w2 = pw3[2];
    float m  = fmaxf(w0, fmaxf(w1, w2));
    float e0 = expf(w0 - m), e1 = expf(w1 - m), e2 = expf(w2 - m);
    float inv = 1.0f / (e0 + e1 + e2);
    int g = t >> 7;
    float cn   = cnt[g];
    float s    = sums[t];
    float mean = s / fmaxf(cn, 1.0f);
    float mxv  = (cn > 0.0f) ? mx[t] : 0.0f;
    pooled[t]  = (s * e0 + mean * e1 + mxv * e2) * inv;
}

// ---------------------------------------------------------------------------
// Head: per-graph block. fcA + LayerNorm + gelu residual + fcB -> [512,64]
// ---------------------------------------------------------------------------
__global__ __launch_bounds__(128)
void head_kernel(const float* __restrict__ pooled,
                 const float* __restrict__ fcAw, const float* __restrict__ fcAb,
                 const float* __restrict__ lng,  const float* __restrict__ lnb,
                 const float* __restrict__ fcBw, const float* __restrict__ fcBb,
                 float* __restrict__ out) {
    __shared__ float row[HID];
    __shared__ float hrow[HID];
    __shared__ float red[HID];
    int g = blockIdx.x, t = threadIdx.x;
    row[t] = pooled[(size_t)g * HID + t];
    __syncthreads();

    float a = fcAb[t];
    for (int k = 0; k < HID; ++k) a += row[k] * fcAw[k * HID + t];

    red[t] = a; __syncthreads();
    for (int s = 64; s > 0; s >>= 1) { if (t < s) red[t] += red[t + s]; __syncthreads(); }
    float mu = red[0] * (1.0f / HID);
    __syncthreads();
    float d = a - mu;
    red[t] = d * d; __syncthreads();
    for (int s = 64; s > 0; s >>= 1) { if (t < s) red[t] += red[t + s]; __syncthreads(); }
    float var = red[0] * (1.0f / HID);

    float o = d * rsqrtf(var + 1e-5f) * lng[t] + lnb[t];
    hrow[t] = gelu_exact(o) + row[t];
    __syncthreads();

    if (t < LAT) {
        float acc = fcBb[t];
        for (int k = 0; k < HID; ++k) acc += hrow[k] * fcBw[k * LAT + t];
        out[(size_t)g * LAT + t] = acc;
    }
}

// ---------------------------------------------------------------------------
// Host orchestration
// ---------------------------------------------------------------------------
extern "C" void kernel_launch(void* const* d_in, const int* in_sizes, int n_in,
                              void* d_out, int out_size, void* d_ws, size_t ws_size,
                              hipStream_t stream) {
    const float* x     = (const float*)d_in[0];
    const int*   edge  = (const int*)d_in[1];
    const int*   batch = (const int*)d_in[2];
    const float* ibn_g = (const float*)d_in[3];
    const float* ibn_b = (const float*)d_in[4];
    const float* eps   = (const float*)d_in[5];
    const float* fc1w  = (const float*)d_in[6];
    // d_in[7] fc1_b: cancelled exactly by following batch-stat BN
    const float* bn1g  = (const float*)d_in[8];
    const float* bn1b  = (const float*)d_in[9];
    const float* fc2w  = (const float*)d_in[10];
    // d_in[11] fc2_b: cancelled by BN
    const float* bng   = (const float*)d_in[12];
    const float* bnb   = (const float*)d_in[13];
    const float* attw  = (const float*)d_in[14];
    const float* poolw = (const float*)d_in[15];
    const float* fcAw  = (const float*)d_in[16];
    const float* fcAb  = (const float*)d_in[17];
    const float* lng   = (const float*)d_in[18];
    const float* lnb   = (const float*)d_in[19];
    const float* fcBw  = (const float*)d_in[20];
    const float* fcBb  = (const float*)d_in[21];
    float* out = (float*)d_out;

    const int* srcI = edge;
    const int* dstI = edge + NEDGES;

    // Workspace carve-out
    char* ws = (char*)d_ws;
    size_t off = 0;
    auto carve = [&](size_t bytes) -> void* {
        void* p = ws + off; off += (bytes + 255) & ~(size_t)255; return p;
    };
    float*  xcur   = (float*) carve((size_t)MPAD * HID * 4);
    float*  agg    = (float*) carve((size_t)MPAD * HID * 4);
    float*  ybuf   = (float*) carve((size_t)MPAD * HID * 4);
    bf16_t* abuf   = (bf16_t*)carve((size_t)MPAD * HID * 2);
    bf16_t* hsbuf  = (bf16_t*)carve((size_t)NLAYERS * MPAD * HID * 2);
    bf16_t* w1t    = (bf16_t*)carve((size_t)NLAYERS * HID * HID * 2);
    bf16_t* w2t    = (bf16_t*)carve((size_t)NLAYERS * HID * HID * 2);
    float*  stats  = (float*) carve(2 * HID * 4);
    float*  ss     = (float*) carve(2 * HID * 4);
    float*  sums   = (float*) carve((size_t)NGRAPHS * HID * 4);
    float*  mx     = (float*) carve((size_t)NGRAPHS * HID * 4);
    float*  cnt    = (float*) carve((size_t)NGRAPHS * 4);
    float*  pooled = (float*) carve((size_t)NGRAPHS * HID * 4);

    const int nElem    = NNODES * HID;                       // real elements
    const int gElem    = (nElem + 255) / 256;
    const int gPadElem = ((MPAD * HID) + 255) / 256;
    const int gStats   = (NNODES + STAT_CHUNK - 1) / STAT_CHUNK;
    const int gGemm    = MPAD / 128;                         // 391
    const float invN   = 1.0f / (float)NNODES;

    // Weight conversion (transposed bf16), every launch (deterministic)
    {
        int tot = NLAYERS * HID * HID;
        wconv_kernel<<<(tot + 255) / 256, 256, 0, stream>>>(fc1w, w1t, NLAYERS);
        wconv_kernel<<<(tot + 255) / 256, 256, 0, stream>>>(fc2w, w2t, NLAYERS);
    }

    // Input BN
    zero_f32_kernel<<<1, 256, 0, stream>>>(stats, 2 * HID);
    colstats_kernel<<<gStats, HID, 0, stream>>>(x, NNODES, stats);
    bn_finalize_kernel<<<1, HID, 0, stream>>>(stats, ibn_g, ibn_b, invN, ss);
    apply_input_bn_kernel<<<gElem, 256, 0, stream>>>(x, ss, xcur, nElem);

    for (int l = 0; l < NLAYERS; ++l) {
        // agg = segment_sum(x[src], dst)
        zero_f32_kernel<<<gPadElem, 256, 0, stream>>>(agg, MPAD * HID);
        scatter_add_kernel<<<(NEDGES * 32 + 255) / 256, 256, 0, stream>>>(
            xcur, srcI, dstI, agg, NEDGES);

        // h_in = (1+eps)*x + agg   -> bf16 A
        hin_kernel<<<gElem, 256, 0, stream>>>(xcur, agg, eps, l, abuf, nElem);

        // GEMM1 (bias cancelled by BN)
        wmma_gemm_kernel<<<gGemm, 256, 0, stream>>>(abuf, w1t + (size_t)l * HID * HID, ybuf);
        zero_f32_kernel<<<1, 256, 0, stream>>>(stats, 2 * HID);
        colstats_kernel<<<gStats, HID, 0, stream>>>(ybuf, NNODES, stats);
        bn_finalize_kernel<<<1, HID, 0, stream>>>(stats, bn1g + l * HID, bn1b + l * HID, invN, ss);
        bn_gelu_kernel<<<gElem, 256, 0, stream>>>(ybuf, ss, abuf, nullptr, nullptr, nElem);

        // GEMM2
        wmma_gemm_kernel<<<gGemm, 256, 0, stream>>>(abuf, w2t + (size_t)l * HID * HID, ybuf);
        zero_f32_kernel<<<1, 256, 0, stream>>>(stats, 2 * HID);
        colstats_kernel<<<gStats, HID, 0, stream>>>(ybuf, NNODES, stats);
        bn_finalize_kernel<<<1, HID, 0, stream>>>(stats, bng + l * HID, bnb + l * HID, invN, ss);
        bn_gelu_kernel<<<gElem, 256, 0, stream>>>(
            ybuf, ss, nullptr, xcur, hsbuf + (size_t)l * MPAD * HID, nElem);
    }

    // JK attention + fused pooling
    pool_init_kernel<<<(NGRAPHS * HID + 255) / 256, 256, 0, stream>>>(sums, mx, cnt);
    jk_pool_kernel<<<(NNODES + 7) / 8, 256, 0, stream>>>(hsbuf, attw, batch, sums, mx, cnt);
    pool_finalize_kernel<<<(NGRAPHS * HID + 255) / 256, 256, 0, stream>>>(
        sums, mx, cnt, poolw, pooled);

    // Head
    head_kernel<<<NGRAPHS, HID, 0, stream>>>(pooled, fcAw, fcAb, lng, lnb, fcBw, fcBb, out);
}
